// EdgeConvConv_90185723281816
// MI455X (gfx1250) — compile-verified, matrix operand order checked by reading, not run on hardware
//
#include <hip/hip_runtime.h>
#include <hip/hip_fp16.h>
#include <cstdint>

typedef __attribute__((ext_vector_type(16))) _Float16 v16h;
typedef __attribute__((ext_vector_type(8)))  _Float16 v8h;
typedef __attribute__((ext_vector_type(8)))  float    v8f;
typedef __attribute__((ext_vector_type(2)))  float    v2f;
typedef __attribute__((ext_vector_type(4)))  unsigned v4u;
typedef __attribute__((ext_vector_type(8)))  int      v8i;
typedef __attribute__((ext_vector_type(4)))  int      v4i;

#define BATCH   8
#define NPTS    4096
#define FEATD   125
#define POSD    3
#define KNN     16
#define DF      128
#define HW      128
#define DOUT    128
#define RADIUS  300.0f
#define BNEPS   1e-5f

#define HAVE_TDM __has_builtin(__builtin_amdgcn_tensor_load_to_lds)

// intra-wave LDS ordering fence (LDS is in-order per wave; this stops compiler
// reordering and drains DScnt)
#define LDS_FENCE() asm volatile("s_wait_dscnt 0" ::: "memory")

// build a v16h fragment from two contiguous 16B LDS chunks (ds_load_b128 x2)
__device__ __forceinline__ v16h frag16(const _Float16* p0, const _Float16* p1) {
  v8h a = *(const v8h*)p0;
  v8h b = *(const v8h*)p1;
  return __builtin_shufflevector(a, b, 0, 1, 2, 3, 4, 5, 6, 7, 8, 9, 10, 11,
                                 12, 13, 14, 15);
}

// ---------------------------------------------------------------------------
// prep kernels
// ---------------------------------------------------------------------------
__global__ void prep_feat_kernel(const float* __restrict__ x,
                                 const float* __restrict__ pos,
                                 float* __restrict__ feat32,
                                 _Float16* __restrict__ feat16) {
  int idx = blockIdx.x * blockDim.x + threadIdx.x;
  const int total = BATCH * NPTS * DF;
  if (idx >= total) return;
  int node = idx >> 7;
  int c    = idx & 127;
  float v = (c < FEATD) ? x[(size_t)node * FEATD + c]
                        : pos[(size_t)node * POSD + (c - FEATD)];
  feat32[idx] = v;
  feat16[idx] = (_Float16)v;
}

__global__ void prep_sq_kernel(const float* __restrict__ feat32,
                               float* __restrict__ sq) {
  int wave = (blockIdx.x * blockDim.x + threadIdx.x) >> 5;
  int lane = threadIdx.x & 31;
  if (wave >= BATCH * NPTS) return;
  const float* f = feat32 + (size_t)wave * DF;
  float s = 0.0f;
  for (int i = lane; i < DF; i += 32) { float v = f[i]; s += v * v; }
  for (int off = 16; off > 0; off >>= 1) s += __shfl_down(s, off);
  if (lane == 0) sq[wave] = s;
}

// fold BN into (scale,bias); pack W1/W2 as f16 WMMA B-fragments
// (lane l, half h: n = l&15, k = ((h<8)? h : 8+h) + ((l&16)? 8 : 0))
__global__ void prep_weights_kernel(
    const float* __restrict__ W1, const float* __restrict__ b1,
    const float* __restrict__ g1, const float* __restrict__ be1,
    const float* __restrict__ m1, const float* __restrict__ v1,
    const float* __restrict__ W2, const float* __restrict__ b2,
    const float* __restrict__ g2, const float* __restrict__ be2,
    const float* __restrict__ m2, const float* __restrict__ v2,
    _Float16* __restrict__ w1pack, _Float16* __restrict__ w2pack,
    float* __restrict__ s1, float* __restrict__ t1,
    float* __restrict__ s2, float* __restrict__ t2) {
  int tid = blockIdx.x * blockDim.x + threadIdx.x;
  int stride = gridDim.x * blockDim.x;

  for (int o = tid; o < HW; o += stride) {
    float s = g1[o] * rsqrtf(v1[o] + BNEPS);
    s1[o] = s;
    t1[o] = (b1[o] - m1[o]) * s + be1[o];
  }
  for (int o = tid; o < DOUT; o += stride) {
    float s = g2[o] * rsqrtf(v2[o] + BNEPS);
    s2[o] = s;
    t2[o] = (b2[o] - m2[o]) * s + be2[o];
  }
  for (int e = tid; e < 8 * 8 * 32 * 16; e += stride) {
    int h    = e & 15;
    int lane = (e >> 4) & 31;
    int sfr  = (e >> 9) & 7;
    int tfr  = (e >> 12) & 7;
    int kk = ((h < 8) ? h : (8 + h)) + ((lane & 16) ? 8 : 0);
    int c = sfr * 32 + kk;
    int o = tfr * 16 + (lane & 15);
    w1pack[e] = (_Float16)W1[(size_t)c * HW + o];
  }
  for (int e = tid; e < 8 * 4 * 32 * 16; e += stride) {
    int h    = e & 15;
    int lane = (e >> 4) & 31;
    int sfr  = (e >> 9) & 3;
    int tfr  = (e >> 11) & 7;
    int kk = ((h < 8) ? h : (8 + h)) + ((lane & 16) ? 8 : 0);
    int c = sfr * 32 + kk;
    int o = tfr * 16 + (lane & 15);
    w2pack[e] = (_Float16)W2[(size_t)c * DOUT + o];
  }
}

// ---------------------------------------------------------------------------
// fused distance + top-K: 8 waves/block share one B tile (TDM double-buffered);
// each wave owns a 16-row tile. Gram via fp32 WMMA (precision for selection).
// ---------------------------------------------------------------------------
#if HAVE_TDM
// 1D TDM descriptor: 2048 f32 contiguous, LDS padded 4 dwords every 128 dwords
// -> LDS row stride of 132 dwords (bank-conflict free fragment reads).
__device__ __forceinline__ void tdm_load_tile(const float* gsrc,
                                              unsigned ldsaddr) {
  unsigned long long ga = (unsigned long long)(uintptr_t)gsrc;
  v4u g0 = {1u,                                    // count=1 valid descriptor
            ldsaddr,                               // lds_addr (bytes)
            (unsigned)ga,                          // global_addr[31:0]
            (unsigned)((ga >> 32) & 0x1ffffffu) | (2u << 30)};  // addr + type=2
  v8i g1 = {(int)((2u << 16)      // data_size = 4B
                  | (1u << 20)    // pad_enable
                  | (6u << 22)    // pad_interval: 128 dwords
                  | (3u << 25)),  // pad_amount: 4 dwords
            (int)(2048u << 16),   // tensor_dim0 = 2048 (low 16 @ bits 63:48)
            (int)(1u << 16),      // tensor_dim0 hi=0, tensor_dim1 = 1
            (int)(2048u << 16),   // tile_dim0 = 2048 @ bits 127:112
            0,                    // tile_dim1=0 (unused), tile_dim2=0
            2048, 0, 0};          // tensor_dim0_stride = 2048
  v4i z  = {0, 0, 0, 0};
  v8i z8 = {0, 0, 0, 0, 0, 0, 0, 0};
  __builtin_amdgcn_tensor_load_to_lds(g0, g1, z, z, z8, 0);
}
#endif

__global__ void __launch_bounds__(256) knn_kernel(
    const float* __restrict__ feat32, const float* __restrict__ sq,
    int* __restrict__ knn_idx, float* __restrict__ knn_dist) {
  const int lane = threadIdx.x & 31;
  const int wid  = threadIdx.x >> 5;
  const int b    = blockIdx.x / (NPTS / 128);
  const int m0   = (blockIdx.x % (NPTS / 128)) * 128 + wid * 16;

  __shared__ __align__(16) float As[8][16][132];
  __shared__ __align__(16) float Bs[2][16][132];
  __shared__ float dt[8][16][16];
  __shared__ float topd[8][16][17];
  __shared__ int   topi[8][16][17];
  __shared__ float rsqs[8][16];
  __shared__ float csq[2][16];

  const float* fb = feat32 + (size_t)b * NPTS * DF;

  { // per-wave A tile (16 x 128 f32, stride-132 rows)
    const float4* src = (const float4*)(fb + (size_t)m0 * DF);
    for (int i = lane; i < 512; i += 32) {
      int r = i >> 5, c4 = i & 31;
      *(float4*)&As[wid][r][c4 * 4] = src[i];
    }
  }
  if (lane < 16) rsqs[wid][lane] = sq[b * NPTS + m0 + lane];
  LDS_FENCE();

  // fp32 A fragments, all 32 k-steps (lo lanes K={4s,4s+1}, hi {4s+2,4s+3})
  v2f af[32];
  {
    const int m  = lane & 15;
    const int ko = (lane & 16) ? 2 : 0;
#pragma unroll
    for (int s = 0; s < 32; ++s) {
      af[s].x = As[wid][m][4 * s + ko];
      af[s].y = As[wid][m][4 * s + ko + 1];
    }
  }

#if HAVE_TDM
  if (wid == 0) tdm_load_tile(fb, (unsigned)(uintptr_t)&Bs[0][0][0]);
#endif

  int cnt = 0, wpos = 0;
  float wmax = -3.4e38f;
  const int nn = lane & 15;
  const int ko = (lane & 16) ? 2 : 0;

  for (int jt = 0; jt < NPTS / 16; ++jt) {
    const int j0 = jt * 16;
#if HAVE_TDM
    const int cur = jt & 1;
    if (wid == 0) {
      if (jt + 1 < NPTS / 16) {
        tdm_load_tile(fb + (size_t)(j0 + 16) * DF,
                      (unsigned)(uintptr_t)&Bs[cur ^ 1][0][0]);
        __builtin_amdgcn_s_wait_tensorcnt(1);  // current tile complete
      } else {
        __builtin_amdgcn_s_wait_tensorcnt(0);
      }
      if (lane < 16) csq[cur][lane] = sq[b * NPTS + j0 + lane];
    }
#else
    const int cur = 0;
    {
      const float4* src = (const float4*)(fb + (size_t)j0 * DF);
      for (int i = threadIdx.x; i < 512; i += 256) {
        int r = i >> 5, c4 = i & 31;
        *(float4*)&Bs[0][r][c4 * 4] = src[i];
      }
      if (threadIdx.x < 16) csq[0][threadIdx.x] = sq[b * NPTS + j0 + threadIdx.x];
      if (j0 + 16 < NPTS)
        __builtin_prefetch(fb + (size_t)(j0 + 16) * DF + threadIdx.x * 8, 0, 1);
    }
#endif
    __syncthreads();  // B tile ready for all waves

    v8f accE = {}, accO = {};
#pragma unroll
    for (int s = 0; s < 32; s += 2) {
      v2f b0, b1;
      b0.x = Bs[cur][nn][4 * s + ko];
      b0.y = Bs[cur][nn][4 * s + ko + 1];
      b1.x = Bs[cur][nn][4 * s + 4 + ko];
      b1.y = Bs[cur][nn][4 * s + 4 + ko + 1];
      accE = __builtin_amdgcn_wmma_f32_16x16x4_f32(false, af[s], false, b0,
                                                   (short)0, accE, false, false);
      accO = __builtin_amdgcn_wmma_f32_16x16x4_f32(false, af[s + 1], false, b1,
                                                   (short)0, accO, false, false);
    }
    v8f acc = accE + accO;

#pragma unroll
    for (int r = 0; r < 8; ++r) {
      int ml = r + ((lane & 16) ? 8 : 0);
      dt[wid][ml][nn] = rsqs[wid][ml] + csq[cur][nn] - 2.0f * acc[r];
    }
    LDS_FENCE();

    if (lane < 16) {  // per-row running top-16 (cheap reject vs current worst)
#pragma unroll 1
      for (int j = 0; j < 16; ++j) {
        float d = dt[wid][lane][j];
        int gi = j0 + j;
        if (cnt < KNN) {
          topd[wid][lane][cnt] = d;
          topi[wid][lane][cnt] = gi;
          if (d > wmax) { wmax = d; wpos = cnt; }
          ++cnt;
        } else if (d < wmax) {
          topd[wid][lane][wpos] = d;
          topi[wid][lane][wpos] = gi;
          wmax = -3.4e38f;
#pragma unroll
          for (int t = 0; t < KNN; ++t) {
            float td = topd[wid][lane][t];
            if (td > wmax) { wmax = td; wpos = t; }
          }
        }
      }
    }
    __syncthreads();  // all waves done with Bs[cur] before it is re-filled
  }

  if (lane < 16) {
    int base = (b * NPTS + m0 + lane) * KNN;
    for (int t = 0; t < KNN; ++t) {
      knn_dist[base + t] = topd[wid][lane][t];
      knn_idx[base + t]  = topi[wid][lane][t];
    }
  }
}

// ---------------------------------------------------------------------------
// fused edge-MLP + masked max-pool: weights staged once per block in LDS,
// 4 waves/block, 8 nodes/wave. M=16 of each WMMA == the K=16 edges.
// ---------------------------------------------------------------------------
#define ENW 4
#define NPW 8

__global__ void __launch_bounds__(128) edge_mlp_kernel(
    const _Float16* __restrict__ feat16,
    const int* __restrict__ knn_idx, const float* __restrict__ knn_dist,
    const _Float16* __restrict__ w1pack, const _Float16* __restrict__ w2pack,
    const float* __restrict__ s1, const float* __restrict__ t1,
    const float* __restrict__ s2, const float* __restrict__ t2,
    float* __restrict__ out) {
  const int lane = threadIdx.x & 31;
  const int wid  = threadIdx.x >> 5;

  __shared__ __align__(16) _Float16 w1l[8 * 8 * 32 * 16];   // 64 KB
  __shared__ __align__(16) _Float16 w2l[8 * 4 * 32 * 16];   // 32 KB
  __shared__ __align__(16) _Float16 ef[ENW][16][136];
  __shared__ __align__(16) _Float16 h1s[ENW][16][136];
  __shared__ __align__(16) _Float16 cfb[ENW][128];
  __shared__ float s1l[128], t1l[128], s2l[128], t2l[128];
  __shared__ float pen[ENW][16];
  __shared__ int   nbr[ENW][16];

  for (int i = threadIdx.x; i < (8 * 8 * 32 * 16) / 8; i += 128)
    ((uint4*)w1l)[i] = ((const uint4*)w1pack)[i];
  for (int i = threadIdx.x; i < (8 * 4 * 32 * 16) / 8; i += 128)
    ((uint4*)w2l)[i] = ((const uint4*)w2pack)[i];
  if (threadIdx.x < 128) {
    int i = threadIdx.x;
    s1l[i] = s1[i]; t1l[i] = t1[i]; s2l[i] = s2[i]; t2l[i] = t2[i];
  }
  __syncthreads();

  const int m   = lane & 15;
  const int nn  = lane & 15;
  const int hi8 = (lane & 16) ? 8 : 0;
  const int node0 = blockIdx.x * (ENW * NPW) + wid * NPW;

  for (int it = 0; it < NPW; ++it) {
    const int node = node0 + it;
    const int b = node / NPTS;

    for (int i = lane; i < 128; i += 32)
      cfb[wid][i] = feat16[(size_t)node * DF + i];
    if (lane < 16) {
      nbr[wid][lane] = knn_idx[node * KNN + lane];
      pen[wid][lane] = (knn_dist[node * KNN + lane] > RADIUS) ? -1.0e8f : 0.0f;
    }
    if (it + 1 < NPW)
      __builtin_prefetch(feat16 + (size_t)(node + 1) * DF + lane * 8, 0, 1);
    LDS_FENCE();

    // gather the 16 neighbor feature rows (L2-resident), 16B chunks
    for (int t = lane; t < 256; t += 32) {
      int r = t >> 4, seg = t & 15;
      int gi = b * NPTS + nbr[wid][r];
      *((uint4*)&ef[wid][r][seg * 8]) =
          ((const uint4*)(feat16 + (size_t)gi * DF))[seg];
    }
    LDS_FENCE();

    // layer-1 A fragments: each lane = two contiguous 8-half runs per frag
    v16h a1[8];
#pragma unroll
    for (int s = 0; s < 8; ++s) {
      int k0 = s * 32 + hi8;
      a1[s] = (s < 4)
                  ? frag16(&cfb[wid][k0], &cfb[wid][k0 + 16])
                  : frag16(&ef[wid][m][k0 - 128], &ef[wid][m][k0 - 128 + 16]);
    }

#pragma unroll
    for (int t = 0; t < 8; ++t) {  // layer 1: [16x256]@[256x128]
      v8f acc0 = {}, acc1 = {};
#pragma unroll
      for (int s = 0; s < 8; s += 2) {
        const _Float16* wp0 = &w1l[((t * 8 + s) * 32 + lane) * 16];
        const _Float16* wp1 = &w1l[((t * 8 + s + 1) * 32 + lane) * 16];
        acc0 = __builtin_amdgcn_wmma_f32_16x16x32_f16(
            false, a1[s], false, frag16(wp0, wp0 + 8), (short)0, acc0, false,
            false);
        acc1 = __builtin_amdgcn_wmma_f32_16x16x32_f16(
            false, a1[s + 1], false, frag16(wp1, wp1 + 8), (short)0, acc1,
            false, false);
      }
      v8f acc = acc0 + acc1;
      int o = t * 16 + nn;
      float sv = s1l[o], tv = t1l[o];
#pragma unroll
      for (int r = 0; r < 8; ++r) {  // folded BN + ReLU
        int mr = r + ((lane & 16) ? 8 : 0);
        h1s[wid][mr][o] = (_Float16)fmaxf(acc[r] * sv + tv, 0.0f);
      }
    }
    LDS_FENCE();

    v16h a2[4];
#pragma unroll
    for (int s = 0; s < 4; ++s) {
      int k0 = s * 32 + hi8;
      a2[s] = frag16(&h1s[wid][m][k0], &h1s[wid][m][k0 + 16]);
    }

#pragma unroll
    for (int t = 0; t < 8; ++t) {  // layer 2: [16x128]@[128x128]
      v8f acc0 = {}, acc1 = {};
#pragma unroll
      for (int s = 0; s < 4; s += 2) {
        const _Float16* wp0 = &w2l[((t * 4 + s) * 32 + lane) * 16];
        const _Float16* wp1 = &w2l[((t * 4 + s + 1) * 32 + lane) * 16];
        acc0 = __builtin_amdgcn_wmma_f32_16x16x32_f16(
            false, a2[s], false, frag16(wp0, wp0 + 8), (short)0, acc0, false,
            false);
        acc1 = __builtin_amdgcn_wmma_f32_16x16x32_f16(
            false, a2[s + 1], false, frag16(wp1, wp1 + 8), (short)0, acc1,
            false, false);
      }
      v8f acc = acc0 + acc1;
      int o = t * 16 + nn;
      float sv = s2l[o], tv = t2l[o];
      float mx = -3.4e38f;
#pragma unroll
      for (int r = 0; r < 8; ++r) {  // folded BN + radius penalty + row-max
        int mr = r + ((lane & 16) ? 8 : 0);
        mx = fmaxf(mx, acc[r] * sv + tv + pen[wid][mr]);
      }
      mx = fmaxf(mx, __shfl_xor(mx, 16));  // merge lane halves (wave32)
      if (lane < 16) out[(size_t)node * DOUT + o] = mx;
    }
  }
}

// ---------------------------------------------------------------------------
extern "C" void kernel_launch(void* const* d_in, const int* in_sizes, int n_in,
                              void* d_out, int out_size, void* d_ws,
                              size_t ws_size, hipStream_t stream) {
  (void)in_sizes; (void)n_in; (void)out_size; (void)ws_size;
  const float* x      = (const float*)d_in[0];
  const float* pos    = (const float*)d_in[1];
  const float* W1     = (const float*)d_in[2];
  const float* b1     = (const float*)d_in[3];
  const float* gamma1 = (const float*)d_in[4];
  const float* beta1  = (const float*)d_in[5];
  const float* mean1  = (const float*)d_in[6];
  const float* var1   = (const float*)d_in[7];
  const float* W2     = (const float*)d_in[8];
  const float* b2     = (const float*)d_in[9];
  const float* gamma2 = (const float*)d_in[10];
  const float* beta2  = (const float*)d_in[11];
  const float* mean2  = (const float*)d_in[12];
  const float* var2   = (const float*)d_in[13];
  float* out = (float*)d_out;

  char* p = (char*)d_ws;
  auto carve = [&p](size_t bytes) {
    void* r = (void*)p;
    p += (bytes + 255) & ~(size_t)255;
    return r;
  };
  float*    feat32   = (float*)carve((size_t)BATCH * NPTS * DF * 4);
  _Float16* feat16   = (_Float16*)carve((size_t)BATCH * NPTS * DF * 2);
  float*    sq       = (float*)carve((size_t)BATCH * NPTS * 4);
  int*      knn_idx  = (int*)carve((size_t)BATCH * NPTS * KNN * 4);
  float*    knn_dist = (float*)carve((size_t)BATCH * NPTS * KNN * 4);
  _Float16* w1pack   = (_Float16*)carve((size_t)8 * 8 * 32 * 16 * 2);
  _Float16* w2pack   = (_Float16*)carve((size_t)8 * 4 * 32 * 16 * 2);
  float*    s1       = (float*)carve(128 * 4);
  float*    t1       = (float*)carve(128 * 4);
  float*    s2       = (float*)carve(128 * 4);
  float*    t2       = (float*)carve(128 * 4);

  const int totalFeat = BATCH * NPTS * DF;
  prep_feat_kernel<<<(totalFeat + 255) / 256, 256, 0, stream>>>(x, pos, feat32,
                                                                feat16);
  prep_sq_kernel<<<(BATCH * NPTS) / 8, 256, 0, stream>>>(feat32, sq);
  prep_weights_kernel<<<64, 256, 0, stream>>>(
      W1, b1, gamma1, beta1, mean1, var1, W2, b2, gamma2, beta2, mean2, var2,
      w1pack, w2pack, s1, t1, s2, t2);
  knn_kernel<<<BATCH * (NPTS / 128), 256, 0, stream>>>(feat32, sq, knn_idx,
                                                       knn_dist);
  edge_mlp_kernel<<<(BATCH * NPTS) / (ENW * NPW), 128, 0, stream>>>(
      feat16, knn_idx, knn_dist, w1pack, w2pack, s1, t1, s2, t2, out);
}
